// RNN_68272800137500
// MI455X (gfx1250) — matrix operand order repaired and checked by reading, not compiled, not run on hardware
//
#include <hip/hip_runtime.h>
#include <hip/hip_bf16.h>

// ---- problem constants ----
#define B_   8
#define T_   50
#define H_   1024
#define P_   1444        // A*A
#define C_   64
#define INX  92416       // A*A*C
#define IN_  93860       // INX + P_
#define KP_  93888       // IN_ padded up to multiple of 64 (bf16 path alignment)
#define G4   4096        // 4*H
#define KSPLIT 16

typedef __attribute__((ext_vector_type(16))) __bf16       v16bf;
typedef __attribute__((ext_vector_type(8)))  float        v8f;
typedef __attribute__((ext_vector_type(2)))  float        f32x2;
typedef __attribute__((ext_vector_type(4)))  unsigned int u32x4;

struct Q2 { u32x4 lo, hi; };   // 32 bytes, bit_cast target for v16bf / v8f

__device__ __forceinline__ float sigm(float x) { return 1.f / (1.f + __expf(-x)); }

// ---------------------------------------------------------------------------
// zero fill (state init each launch; harness poisons d_ws)
// ---------------------------------------------------------------------------
__global__ void zero_f(float* __restrict__ p, size_t n) {
  size_t i = (size_t)blockIdx.x * blockDim.x + threadIdx.x;
  if (i < n) p[i] = 0.f;
}

// ---------------------------------------------------------------------------
// one-time fp32 -> bf16 weight conversion with K padding (pad = 0)
// dst layout: [rows][kdst]
// ---------------------------------------------------------------------------
__global__ void convert_pad(const float* __restrict__ src, __hip_bfloat16* __restrict__ dst,
                            int ksrc, int kdst) {
  int k = blockIdx.x * 256 + threadIdx.x;
  int r = blockIdx.y;
  if (k >= kdst) return;
  float v = (k < ksrc) ? src[(size_t)r * ksrc + k] : 0.f;
  dst[(size_t)r * kdst + k] = __float2bfloat16(v);
}

// ---------------------------------------------------------------------------
// build inp[b][k] = concat( x_t * att (p-major, c-minor), m_t ), pad with 0
// ---------------------------------------------------------------------------
template <typename T>
__global__ void build_inp(const float* __restrict__ tensors, const float* __restrict__ masks,
                          const float* __restrict__ att, T* __restrict__ inp, int t, int ld) {
  int k = blockIdx.x * 256 + threadIdx.x;
  int b = blockIdx.y;
  if (k >= ld) return;
  float v = 0.f;
  if (k < INX) {
    int p = k >> 6, cc = k & 63;
    v = tensors[(((size_t)b * T_ + t) * P_ + p) * C_ + cc] * att[b * P_ + p];
  } else if (k < IN_) {
    v = masks[((size_t)b * T_ + t) * P_ + (k - INX)];
  }
  inp[(size_t)b * ld + k] = (T)v;
}

// ---------------------------------------------------------------------------
// Split-K GEMM, bf16 WMMA:  partial[chunkBase+by][j][b] = sum_k A[b,k]*W[j,k]
// A: [16][ldA] bf16 (rows 8..15 are zero), W: [4096][ldW] bf16
// One wave per block; block.x -> 16-wide j tile; block.y -> K chunk.
// ---------------------------------------------------------------------------
__global__ __launch_bounds__(32) void gemm_bf16_wmma(
    const __hip_bfloat16* __restrict__ A, const __hip_bfloat16* __restrict__ W,
    int ldA, int ldW, int kEnd, int chunkK,
    float* __restrict__ partial, int chunkBase) {
  const int lane  = threadIdx.x;
  const int half  = lane >> 4;      // 0|1
  const int l15   = lane & 15;
  const int jbase = blockIdx.x * 16;
  const int chunk = blockIdx.y;
  int k0 = chunk * chunkK;
  int k1 = k0 + chunkK; if (k1 > kEnd) k1 = kEnd;

  // B operand: lane covers column j=jbase+l15, K range [k+16*half, k+16*half+16)
  const __hip_bfloat16* wp = W + (size_t)(jbase + l15) * ldW + (half << 4);
  // A operand: lane covers row M=l15, K range [k+16*half, k+16*half+16)
  const __hip_bfloat16* ap = A + (size_t)l15 * ldA + (half << 4);

  v8f acc = {0.f, 0.f, 0.f, 0.f, 0.f, 0.f, 0.f, 0.f};
#pragma unroll 2
  for (int k = k0; k < k1; k += 32) {
    v16bf bm = __builtin_nontemporal_load((const v16bf*)(wp + k));  // streamed weights: NT
    Q2 aq;
    aq.lo = *(const u32x4*)(ap + k);
    aq.hi = *(const u32x4*)(ap + k + 8);
    v16bf am = __builtin_bit_cast(v16bf, aq);
    acc = __builtin_amdgcn_wmma_f32_16x16x32_bf16(false, am, false, bm, (short)0, acc,
                                                  false, false);
  }
  if (lane < 16) {  // lanes 0-15: VGPR r = (batch r, j = jbase+lane); batches 8-15 discarded
    float* p = partial + ((size_t)(chunkBase + chunk) * G4 + jbase + lane) * 8;
    Q2 s = __builtin_bit_cast(Q2, acc);
    *(u32x4*)(p)     = s.lo;
    *(u32x4*)(p + 4) = s.hi;
  }
}

// ---------------------------------------------------------------------------
// Split-K GEMM, fp32 WMMA (fallback when ws can't hold bf16 weight image)
// ---------------------------------------------------------------------------
__global__ __launch_bounds__(32) void gemm_f32_wmma(
    const float* __restrict__ A, const float* __restrict__ W,
    int ldA, int ldW, int kEnd, int chunkK,
    float* __restrict__ partial, int chunkBase) {
  const int lane  = threadIdx.x;
  const int half  = lane >> 4;
  const int l15   = lane & 15;
  const int jbase = blockIdx.x * 16;
  const int chunk = blockIdx.y;
  int k0 = chunk * chunkK;
  int k1 = k0 + chunkK; if (k1 > kEnd) k1 = kEnd;

  const float* wp = W + (size_t)(jbase + l15) * ldW + (half << 1);  // B: K = 2*half+{0,1}
  const float* ap = A + (size_t)l15 * ldA + (half << 1);            // A: K = 2*half+{0,1}

  v8f acc = {0.f, 0.f, 0.f, 0.f, 0.f, 0.f, 0.f, 0.f};
#pragma unroll 4
  for (int k = k0; k < k1; k += 4) {
    f32x2 bm = __builtin_nontemporal_load((const f32x2*)(wp + k));
    f32x2 am = *(const f32x2*)(ap + k);
    acc = __builtin_amdgcn_wmma_f32_16x16x4_f32(false, am, false, bm, (short)0, acc,
                                                false, false);
  }
  if (lane < 16) {
    float* p = partial + ((size_t)(chunkBase + chunk) * G4 + jbase + lane) * 8;
    Q2 s = __builtin_bit_cast(Q2, acc);
    *(u32x4*)(p)     = s.lo;
    *(u32x4*)(p + 4) = s.hi;
  }
}

// ---------------------------------------------------------------------------
// reduce split-K partials + biases, apply LSTM pointwise, emit h / c / h_bf16 / relu(h)
// thread idx = j*8 + b  (b fastest -> coalesced partial reads)
// ---------------------------------------------------------------------------
__global__ __launch_bounds__(256) void lstm_reduce(
    const float* __restrict__ partial, int nchunks,
    const float* __restrict__ b_ih, const float* __restrict__ b_hh,
    float* __restrict__ h, float* __restrict__ c,
    __hip_bfloat16* __restrict__ hbf, float* __restrict__ hrelu) {
  int idx = blockIdx.x * 256 + threadIdx.x;  // 0..8191
  int b = idx & 7, j = idx >> 3;
  float g[4];
#pragma unroll
  for (int q = 0; q < 4; ++q) {            // torch gate order: i, f, g, o
    int col = q * H_ + j;
    float s = b_ih[col] + b_hh[col];
    for (int ch = 0; ch < nchunks; ++ch)
      s += partial[((size_t)ch * G4 + col) * 8 + b];
    g[q] = s;
  }
  float ig = sigm(g[0]), fg = sigm(g[1]), gg = tanhf(g[2]), og = sigm(g[3]);
  int ci = b * H_ + j;
  float cn = fg * c[ci] + ig * gg;
  float hn = og * tanhf(cn);
  c[ci]     = cn;
  h[ci]     = hn;
  hbf[ci]   = __float2bfloat16(hn);
  hrelu[ci] = fmaxf(hn, 0.f);
}

// ---------------------------------------------------------------------------
// att = sigmoid(relu(h) @ W_att^T + b_att);  out_t = sigmoid(relu(h) @ W_out^T + b_out)
// h_relu cached in LDS; one wave per output row p; shfl_xor wave32 reduction.
// ---------------------------------------------------------------------------
__global__ __launch_bounds__(256) void att_out(
    const float* __restrict__ hrelu, const float* __restrict__ W_att,
    const float* __restrict__ b_att, const float* __restrict__ W_out,
    const float* __restrict__ b_out, float* __restrict__ att,
    float* __restrict__ out, int t) {
  __shared__ float hs[8 * H_];  // 32 KB
  for (int i = threadIdx.x; i < 8 * H_; i += 256) hs[i] = hrelu[i];
  __syncthreads();

  int wave = threadIdx.x >> 5, lane = threadIdx.x & 31;
  int p = blockIdx.x * 8 + wave;
  if (p >= P_) return;

  const float* wa = W_att + (size_t)p * H_;
  const float* wo = W_out + (size_t)p * H_;
  float aA[8] = {0.f, 0.f, 0.f, 0.f, 0.f, 0.f, 0.f, 0.f};
  float aO[8] = {0.f, 0.f, 0.f, 0.f, 0.f, 0.f, 0.f, 0.f};
  for (int k = lane; k < H_; k += 32) {
    float va = wa[k], vo = wo[k];
#pragma unroll
    for (int b = 0; b < 8; ++b) {
      float hv = hs[b * H_ + k];
      aA[b] += va * hv;
      aO[b] += vo * hv;
    }
  }
#pragma unroll
  for (int off = 16; off > 0; off >>= 1) {
#pragma unroll
    for (int b = 0; b < 8; ++b) {
      aA[b] += __shfl_xor(aA[b], off, 32);
      aO[b] += __shfl_xor(aO[b], off, 32);
    }
  }
  if (lane == 0) {
#pragma unroll
    for (int b = 0; b < 8; ++b) {
      att[b * P_ + p] = sigm(aA[b] + b_att[p]);
      out[((size_t)b * T_ + t) * P_ + p] = sigm(aO[b] + b_out[p]);
    }
  }
}

// ---------------------------------------------------------------------------
extern "C" void kernel_launch(void* const* d_in, const int* in_sizes, int n_in,
                              void* d_out, int out_size, void* d_ws, size_t ws_size,
                              hipStream_t stream) {
  const float* tensors = (const float*)d_in[0];
  const float* masks   = (const float*)d_in[1];
  const float* W_ih    = (const float*)d_in[2];
  const float* W_hh    = (const float*)d_in[3];
  const float* b_ih    = (const float*)d_in[4];
  const float* b_hh    = (const float*)d_in[5];
  const float* W_att   = (const float*)d_in[6];
  const float* b_att   = (const float*)d_in[7];
  const float* W_out   = (const float*)d_in[8];
  const float* b_out   = (const float*)d_in[9];
  float* out = (float*)d_out;

  // workspace layout (256 B aligned slabs)
  char* ws = (char*)d_ws;
  size_t off = 0;
  auto alloc = [&](size_t bytes) { size_t o = off; off = (off + bytes + 255) & ~(size_t)255; return o; };
  size_t o_h    = alloc((size_t)16 * H_ * 4);   // h fp32, rows 8..15 stay zero (WMMA M-pad)
  size_t o_c    = alloc((size_t)8 * H_ * 4);
  size_t o_att  = alloc((size_t)8 * P_ * 4);
  size_t o_hbf  = alloc((size_t)16 * H_ * 2);   // h bf16, rows 8..15 zero
  size_t o_hrel = alloc((size_t)8 * H_ * 4);
  size_t o_inp  = alloc((size_t)16 * KP_ * 4);  // inp, sized for fp32; bf16 path uses half
  size_t stateEnd = off;                        // everything above zero-inited each launch
  size_t o_part = alloc((size_t)(KSPLIT + 1) * G4 * 8 * 4);
  size_t o_wih  = alloc((size_t)G4 * KP_ * 2);  // bf16 weight image (bf16 path only)
  size_t o_whh  = alloc((size_t)G4 * H_ * 2);
  size_t bf16End = off;
  bool use_bf16 = (ws_size >= bf16End);         // deterministic: depends only on ws_size

  float*          h     = (float*)(ws + o_h);
  float*          c     = (float*)(ws + o_c);
  float*          att   = (float*)(ws + o_att);
  __hip_bfloat16* hbf   = (__hip_bfloat16*)(ws + o_hbf);
  float*          hrel  = (float*)(ws + o_hrel);
  float*          inpF  = (float*)(ws + o_inp);
  __hip_bfloat16* inpB  = (__hip_bfloat16*)(ws + o_inp);
  float*          part  = (float*)(ws + o_part);
  __hip_bfloat16* wihB  = (__hip_bfloat16*)(ws + o_wih);
  __hip_bfloat16* whhB  = (__hip_bfloat16*)(ws + o_whh);

  // init recurrent state (h=c=att=0) and zero padded rows of inp / h buffers
  {
    size_t zn = stateEnd / 4;
    zero_f<<<(unsigned)((zn + 255) / 256), 256, 0, stream>>>((float*)ws, zn);
  }

  // one-time (per launch) bf16 weight image: halves the 50x W_ih HBM stream
  if (use_bf16) {
    convert_pad<<<dim3((KP_ + 255) / 256, G4), 256, 0, stream>>>(W_ih, wihB, IN_, KP_);
    convert_pad<<<dim3((H_ + 255) / 256, G4), 256, 0, stream>>>(W_hh, whhB, H_, H_);
  }

  // split-K chunk sizes (bf16: multiple of 32; fp32: multiple of 4)
  const int chunkB = ((KP_ + KSPLIT * 32 - 1) / (KSPLIT * 32)) * 32;  // 5888
  const int chunkF = ((IN_ + KSPLIT * 4 - 1) / (KSPLIT * 4)) * 4;     // 5868

  for (int t = 0; t < T_; ++t) {
    if (use_bf16) {
      build_inp<__hip_bfloat16><<<dim3((KP_ + 255) / 256, B_), 256, 0, stream>>>(
          tensors, masks, att, inpB, t, KP_);
      gemm_bf16_wmma<<<dim3(G4 / 16, KSPLIT), 32, 0, stream>>>(
          inpB, wihB, KP_, KP_, KP_, chunkB, part, 0);
      gemm_bf16_wmma<<<dim3(G4 / 16, 1), 32, 0, stream>>>(
          hbf, whhB, H_, H_, H_, H_, part, KSPLIT);
    } else {
      build_inp<float><<<dim3((IN_ + 255) / 256, B_), 256, 0, stream>>>(
          tensors, masks, att, inpF, t, IN_);
      gemm_f32_wmma<<<dim3(G4 / 16, KSPLIT), 32, 0, stream>>>(
          inpF, W_ih, IN_, IN_, IN_, chunkF, part, 0);
      gemm_f32_wmma<<<dim3(G4 / 16, 1), 32, 0, stream>>>(
          h, W_hh, H_, H_, H_, H_, part, KSPLIT);
    }
    lstm_reduce<<<(B_ * H_) / 256, 256, 0, stream>>>(
        part, KSPLIT + 1, b_ih, b_hh, h, c, hbf, hrel);
    att_out<<<(P_ + 7) / 8, 256, 0, stream>>>(
        hrel, W_att, b_att, W_out, b_out, att, out, t);
  }
  (void)in_sizes; (void)n_in; (void)out_size;
}
